// GatedSelfAttention_64287070486789
// MI455X (gfx1250) — matrix operand order repaired and checked by reading, not compiled, not run on hardware
//
#include <hip/hip_runtime.h>
#include <hip/hip_bf16.h>
#include <cstdint>
#include <cstddef>

// Problem constants (match reference: B=4, S=2048, H=1024)
static constexpr int Bn = 4;
static constexpr int Sn = 2048;
static constexpr int Hn = 1024;

typedef __attribute__((ext_vector_type(16))) __bf16 v16bf;
typedef __attribute__((ext_vector_type(8)))  __bf16 v8bf;
typedef __attribute__((ext_vector_type(8)))  float  v8f;

// ---------- bf16 helpers (RNE convert) ----------
__device__ __forceinline__ __bf16 f2bf(float f) {
    unsigned u = __builtin_bit_cast(unsigned, f);
    unsigned r = u + 0x7FFFu + ((u >> 16) & 1u);
    unsigned short h = (unsigned short)(r >> 16);
    return __builtin_bit_cast(__bf16, h);
}
__device__ __forceinline__ float bf2f(__bf16 b) {
    unsigned short h = __builtin_bit_cast(unsigned short, b);
    unsigned u = ((unsigned)h) << 16;
    return __builtin_bit_cast(float, u);
}

// Build a 16-element bf16 fragment from two contiguous 8-element (16B) runs.
__device__ __forceinline__ v16bf load_frag(const __bf16* lo, const __bf16* hi) {
    v8bf a = *reinterpret_cast<const v8bf*>(lo);
    v8bf b = *reinterpret_cast<const v8bf*>(hi);
    v16bf r;
#pragma unroll
    for (int i = 0; i < 8; ++i) { r[i] = a[i]; r[i + 8] = b[i]; }
    return r;
}

// ---------- conversion kernels ----------
__global__ void k_f2bf(const float* __restrict__ src, __bf16* __restrict__ dst, int n) {
    for (int i = blockIdx.x * blockDim.x + threadIdx.x; i < n; i += gridDim.x * blockDim.x)
        dst[i] = f2bf(src[i]);
}

// Wt[d][h] = bf16(W[h][d])  (so B-operand rows are contiguous along K)
__global__ void k_wT(const float* __restrict__ W, __bf16* __restrict__ Wt) {
    int i = blockIdx.x * blockDim.x + threadIdx.x;
    if (i < Hn * Hn) {
        int d = i / Hn, h = i % Hn;
        Wt[i] = f2bf(W[(size_t)h * Hn + d]);
    }
}

// ---------- WMMA GEMM:  out[rows][N] = A[rows][K] * Wt[N][K]^T + bias ----------
// 8 waves per block; each wave owns a 64x64 output macro-tile (4x4 WMMA tiles),
// so each k-step issues 16 WMMAs for 8 fragment loads (2 wmma / fragment).
template <bool F32OUT, bool TRANSOUT>
__global__ void __launch_bounds__(256)
gemm_bf16_wmma(const __bf16* __restrict__ A, const __bf16* __restrict__ Wt,
               const float* __restrict__ bias,
               __bf16* __restrict__ outb, float* __restrict__ outf,
               int rows, int N, int K) {
    const int lane = threadIdx.x & 31;
    const int wv   = threadIdx.x >> 5;
    const int tile = blockIdx.x * 8 + wv;
    const int tilesN = N >> 6;                      // 64-wide macro tiles
    const int tM = tile / tilesN, tN = tile % tilesN;
    if (tM * 64 >= rows) return;
    const int half = (lane >> 4) & 1, l16 = lane & 15;

    const __bf16* arow[4];
    const __bf16* brow[4];
#pragma unroll
    for (int i = 0; i < 4; ++i) {
        arow[i] = A  + (size_t)(tM * 64 + i * 16 + l16) * K;   // A frag: M = l16
        brow[i] = Wt + (size_t)(tN * 64 + i * 16 + l16) * K;   // B frag: N = l16
    }

    v8f c[4][4];
#pragma unroll
    for (int mr = 0; mr < 4; ++mr)
#pragma unroll
        for (int nr = 0; nr < 4; ++nr) c[mr][nr] = {};

    for (int k0 = 0; k0 < K; k0 += 32) {
        // stream next K-panel into cache (global_prefetch_b8), once per 4 k-steps
        if ((k0 & 127) == 0 && k0 + 128 < K) {
#pragma unroll
            for (int i = 0; i < 4; ++i) {
                __builtin_prefetch(arow[i] + k0 + 128, 0, 3);
                __builtin_prefetch(brow[i] + k0 + 128, 0, 3);
            }
        }
        v16bf a[4], b[4];
#pragma unroll
        for (int i = 0; i < 4; ++i) {
            // A 16x32 bf16 layout: e<8 -> K = e + 8*half ; e>=8 -> K = e + 8 + 8*half
            a[i] = load_frag(arow[i] + k0 + half * 8, arow[i] + k0 + 16 + half * 8);
            // B 32x16 bf16 layout: e<8 -> K = e + 8*half ; e>=8 -> K = e + 8 + 8*half
            b[i] = load_frag(brow[i] + k0 + half * 8, brow[i] + k0 + 16 + half * 8);
        }
#pragma unroll
        for (int mr = 0; mr < 4; ++mr)
#pragma unroll
            for (int nr = 0; nr < 4; ++nr)
                c[mr][nr] = __builtin_amdgcn_wmma_f32_16x16x32_bf16(
                    false, a[mr], false, b[nr], (short)0, c[mr][nr], false, false);
    }

#pragma unroll
    for (int nr = 0; nr < 4; ++nr) {
        const int n = tN * 64 + nr * 16 + l16;
        const float bv = bias ? bias[n] : 0.0f;
#pragma unroll
        for (int mr = 0; mr < 4; ++mr) {
#pragma unroll
            for (int r = 0; r < 8; ++r) {           // C/D: M = r + 8*half, N = l16
                const int m = tM * 64 + mr * 16 + r + half * 8;
                const float v = c[mr][nr][r] + bv;
                if (F32OUT)        outf[(size_t)m * N + n] = v;
                else if (TRANSOUT) outb[(size_t)n * rows + m] = f2bf(v);  // V^T
                else               outb[(size_t)m * N + n] = f2bf(v);
            }
        }
    }
}

// ---------- flash attention + gate ----------
// Block = 256 threads (8 waves) owns 16 queries x full H. Wave w owns dims [w*128, w*128+128).
__global__ void __launch_bounds__(256)
attn_flash_wmma(const __bf16* __restrict__ Qb, const __bf16* __restrict__ Kb,
                const __bf16* __restrict__ VbT, const __bf16* __restrict__ Gb,
                const int* __restrict__ mask, __bf16* __restrict__ AGb) {
    __shared__ __align__(16) float  s_scores[16 * 32];
    __shared__ __align__(16) __bf16 s_p[16 * 32];
    __shared__ float s_alpha[16];
    __shared__ float s_linv[16];

    const int tid  = threadIdx.x;
    const int lane = tid & 31, wv = tid >> 5;
    const int half = (lane >> 4) & 1, l16 = lane & 15;
    const int qtPer = Sn / 16;
    const int b  = blockIdx.x / qtPer;
    const int q0 = (blockIdx.x % qtPer) * 16;
    const size_t rowbase = (size_t)b * Sn + q0;
    const float scale = 0.03125f;  // 1/sqrt(1024)

    // Preload this wave's Q fragments (16 queries x 128 dims = 4 k-steps)
    v16bf qf[4];
    {
        const __bf16* qrow = Qb + (rowbase + l16) * Hn + wv * 128;
#pragma unroll
        for (int kk = 0; kk < 4; ++kk)
            qf[kk] = load_frag(qrow + kk * 32 + half * 8, qrow + kk * 32 + 16 + half * 8);
    }

    v8f acc[8];
#pragma unroll
    for (int i = 0; i < 8; ++i) acc[i] = {};
    float m_run = -__builtin_inff(), l_run = 0.0f;

    for (int kb = 0; kb < Sn; kb += 32) {
        // zero the shared score tile
        for (int i = tid; i < 16 * 32; i += 256) s_scores[i] = 0.0f;
        __syncthreads();

        // partial scores over this wave's 128-dim slice, reduced via LDS f32 atomics
#pragma unroll
        for (int nt = 0; nt < 2; ++nt) {
            v8f sc = {};
            const __bf16* krow = Kb + ((size_t)b * Sn + kb + nt * 16 + l16) * Hn + wv * 128;
#pragma unroll
            for (int kk = 0; kk < 4; ++kk) {
                v16bf kf = load_frag(krow + kk * 32 + half * 8,
                                     krow + kk * 32 + 16 + half * 8);
                sc = __builtin_amdgcn_wmma_f32_16x16x32_bf16(false, qf[kk], false, kf,
                                                             (short)0, sc, false, false);
            }
#pragma unroll
            for (int r = 0; r < 8; ++r) {
                int M = r + half * 8, N = nt * 16 + l16;
                atomicAdd(&s_scores[M * 32 + N], sc[r]);
            }
        }
        __syncthreads();

        // wave 0, lanes 0..15: online softmax for row = tid
        if (tid < 16) {
            const int row = tid;
            const int* mrow = mask + (size_t)(q0 + row) * Sn + kb;
            float s[32];
            float mx = m_run;
#pragma unroll
            for (int j = 0; j < 32; ++j) {
                float v = s_scores[row * 32 + j] * scale;
                if (mrow[j] == 0) v = -__builtin_inff();
                s[j] = v;
                mx = fmaxf(mx, v);
            }
            float alpha, psum = 0.0f;
            if (mx == -__builtin_inff()) {
                alpha = 1.0f;
#pragma unroll
                for (int j = 0; j < 32; ++j) s_p[row * 32 + j] = f2bf(0.0f);
            } else {
                alpha = (m_run == -__builtin_inff()) ? 0.0f : __expf(m_run - mx);
#pragma unroll
                for (int j = 0; j < 32; ++j) {
                    float p = __expf(s[j] - mx);
                    psum += p;
                    s_p[row * 32 + j] = f2bf(p);
                }
            }
            l_run = l_run * alpha + psum;
            m_run = mx;
            s_alpha[row] = alpha;
        }
        __syncthreads();

        // all waves: rescale accumulators, then acc += P(16x32) @ V(32x16-slices)
        float al[8];
#pragma unroll
        for (int r = 0; r < 8; ++r) al[r] = s_alpha[r + half * 8];
        v16bf pf = load_frag(&s_p[l16 * 32 + half * 8], &s_p[l16 * 32 + 16 + half * 8]);
#pragma unroll
        for (int nt = 0; nt < 8; ++nt) {
#pragma unroll
            for (int r = 0; r < 8; ++r) acc[nt][r] *= al[r];
            int col = wv * 128 + nt * 16 + l16;
            const __bf16* vcol = VbT + (size_t)col * (Bn * Sn) + (size_t)b * Sn + kb;
            v16bf vf = load_frag(vcol + half * 8, vcol + 16 + half * 8);
            acc[nt] = __builtin_amdgcn_wmma_f32_16x16x32_bf16(false, pf, false, vf,
                                                              (short)0, acc[nt],
                                                              false, false);
        }
        __syncthreads();
    }

    // finalize: divide by l, multiply by sigmoid gate, store bf16
    if (tid < 16) s_linv[tid] = (l_run > 0.0f) ? 1.0f / l_run : 0.0f;
    __syncthreads();
    float li[8];
#pragma unroll
    for (int r = 0; r < 8; ++r) li[r] = s_linv[r + half * 8];
#pragma unroll
    for (int nt = 0; nt < 8; ++nt) {
        int col = wv * 128 + nt * 16 + l16;
#pragma unroll
        for (int r = 0; r < 8; ++r) {
            size_t row = rowbase + r + half * 8;
            float g = bf2f(Gb[row * Hn + col]);
            g = 1.0f / (1.0f + __expf(-g));
            AGb[row * Hn + col] = f2bf(acc[nt][r] * li[r] * g);
        }
    }
}

// ---------- host launcher ----------
extern "C" void kernel_launch(void* const* d_in, const int* in_sizes, int n_in,
                              void* d_out, int out_size, void* d_ws, size_t ws_size,
                              hipStream_t stream) {
    const float* X    = (const float*)d_in[0];
    const int*   mask = (const int*)d_in[1];
    const float* Wq = (const float*)d_in[2];  const float* bq = (const float*)d_in[3];
    const float* Wk = (const float*)d_in[4];  const float* bk = (const float*)d_in[5];
    const float* Wv = (const float*)d_in[6];  const float* bv = (const float*)d_in[7];
    const float* Wg = (const float*)d_in[8];  const float* bg = (const float*)d_in[9];
    const float* Wo = (const float*)d_in[10]; const float* bo = (const float*)d_in[11];
    float* out = (float*)d_out;

    const size_t nBS = (size_t)Bn * Sn;       // 8192 rows
    char* p = (char*)d_ws;
    __bf16* Xb  = (__bf16*)p; p += nBS * Hn * sizeof(__bf16);
    __bf16* Wtq = (__bf16*)p; p += (size_t)Hn * Hn * sizeof(__bf16);
    __bf16* Wtk = (__bf16*)p; p += (size_t)Hn * Hn * sizeof(__bf16);
    __bf16* Wtv = (__bf16*)p; p += (size_t)Hn * Hn * sizeof(__bf16);
    __bf16* Wtg = (__bf16*)p; p += (size_t)Hn * Hn * sizeof(__bf16);
    __bf16* Wto = (__bf16*)p; p += (size_t)Hn * Hn * sizeof(__bf16);
    __bf16* Qb  = (__bf16*)p; p += nBS * Hn * sizeof(__bf16);
    __bf16* Kb  = (__bf16*)p; p += nBS * Hn * sizeof(__bf16);
    __bf16* VbT = (__bf16*)p; p += nBS * Hn * sizeof(__bf16);   // [H][B*S]
    __bf16* Gb  = (__bf16*)p; p += nBS * Hn * sizeof(__bf16);
    __bf16* AGb = (__bf16*)p; p += nBS * Hn * sizeof(__bf16);
    (void)ws_size; (void)in_sizes; (void)n_in; (void)out_size;  // ~106 MB used

    // 1) precision conversion / weight transpose
    k_f2bf<<<2048, 256, 0, stream>>>(X, Xb, (int)(nBS * Hn));
    const int wBlocks = (Hn * Hn + 255) / 256;
    k_wT<<<wBlocks, 256, 0, stream>>>(Wq, Wtq);
    k_wT<<<wBlocks, 256, 0, stream>>>(Wk, Wtk);
    k_wT<<<wBlocks, 256, 0, stream>>>(Wv, Wtv);
    k_wT<<<wBlocks, 256, 0, stream>>>(Wg, Wtg);
    k_wT<<<wBlocks, 256, 0, stream>>>(Wo, Wto);

    // 2) projections (bf16 out); V stored transposed for the P@V inner loop
    const int gemmBlocks = (int)((nBS / 64) * (Hn / 64) / 8);   // 256
    gemm_bf16_wmma<false, false><<<gemmBlocks, 256, 0, stream>>>(
        Xb, Wtq, bq, Qb, nullptr, (int)nBS, Hn, Hn);
    gemm_bf16_wmma<false, false><<<gemmBlocks, 256, 0, stream>>>(
        Xb, Wtk, bk, Kb, nullptr, (int)nBS, Hn, Hn);
    gemm_bf16_wmma<false, true><<<gemmBlocks, 256, 0, stream>>>(
        Xb, Wtv, bv, VbT, nullptr, (int)nBS, Hn, Hn);
    gemm_bf16_wmma<false, false><<<gemmBlocks, 256, 0, stream>>>(
        Xb, Wtg, bg, Gb, nullptr, (int)nBS, Hn, Hn);

    // 3) flash attention + sigmoid gate -> bf16
    attn_flash_wmma<<<(int)(nBS / 16), 256, 0, stream>>>(Qb, Kb, VbT, Gb, mask, AGb);

    // 4) output projection, fp32 straight to d_out
    gemm_bf16_wmma<true, false><<<gemmBlocks, 256, 0, stream>>>(
        AGb, Wto, bo, nullptr, out, (int)nBS, Hn, Hn);
}
